// AttentionHead_42460046688769
// MI455X (gfx1250) — matrix operand order repaired and checked by reading, not compiled, not run on hardware
//
#include <hip/hip_runtime.h>

#define USE_TDM 1

typedef __attribute__((ext_vector_type(16))) _Float16 v16h;
typedef __attribute__((ext_vector_type(8)))  float    v8f;
typedef __attribute__((ext_vector_type(4)))  unsigned int uint32x4;
typedef __attribute__((ext_vector_type(8)))  int          int32x8;

#define WMMA_F32_F16(a, b, c) \
  __builtin_amdgcn_wmma_f32_16x16x32_f16(false, (a), false, (b), (short)0, (c), false, false)

static constexpr int Bn = 4, Cn = 64, Dn = 64, Nn = 4096;

__device__ __forceinline__ float red16_max(float v) {
  v = fmaxf(v, __shfl_xor(v, 1, 32));
  v = fmaxf(v, __shfl_xor(v, 2, 32));
  v = fmaxf(v, __shfl_xor(v, 4, 32));
  v = fmaxf(v, __shfl_xor(v, 8, 32));
  return v;
}
__device__ __forceinline__ float red16_sum(float v) {
  v += __shfl_xor(v, 1, 32);
  v += __shfl_xor(v, 2, 32);
  v += __shfl_xor(v, 4, 32);
  v += __shfl_xor(v, 8, 32);
  return v;
}

#if USE_TDM
// Issue a TDM 2-D tile load (Global -> LDS), 16-bit elements.
// D# per CDNA5 ISA ch.8: group0 = {count, lds_addr, global_addr, type=2},
// group1 = {data_size=2B, tensor_dim0/1, tile_dim0/1/2, tensor_dim0_stride}.
// 2-operand form => VADDR2/3 = NULL (tensor rank <= 2).
__device__ __forceinline__ void tdm_load_2d_f16(unsigned lds_addr, const void* gptr,
                                                unsigned dim0, unsigned dim1,
                                                unsigned tile0, unsigned tile1,
                                                unsigned stride0) {
  unsigned long long ga = (unsigned long long)gptr;
  uint32x4 g0;
  g0[0] = 1u;                                               // count=1, user D#
  g0[1] = lds_addr;                                         // LDS byte address
  g0[2] = (unsigned)ga;                                     // global addr [31:0]
  g0[3] = (unsigned)((ga >> 32) & 0x01FFFFFFu) | (2u << 30); // addr[56:32] | type=2
  int32x8 g1;
  g1[0] = (int)(1u << 16);                                  // wg_mask=0, data_size=2B
  g1[1] = (int)((dim0 & 0xFFFFu) << 16);                    // tensor_dim0[15:0]
  g1[2] = (int)(((dim0 >> 16) & 0xFFFFu) | ((dim1 & 0xFFFFu) << 16));
  g1[3] = (int)(((dim1 >> 16) & 0xFFFFu) | ((tile0 & 0xFFFFu) << 16));
  g1[4] = (int)(tile1 & 0xFFFFu);                           // tile_dim1; tile_dim2=0
  g1[5] = (int)stride0;                                     // dim0_stride[31:0]
  g1[6] = 0;                                                // stride hi / dim1_stride lo
  g1[7] = 0;
  asm volatile("tensor_load_to_lds %0, %1" :: "s"(g0), "s"(g1) : "memory");
}
#endif

// ---------------------------------------------------------------------------
// Kernel 1: InstanceNorm (GroupNorm w/ groups==C) -> f16 xn[B][C][N]
// ---------------------------------------------------------------------------
__global__ __launch_bounds__(256) void k_inorm(const float* __restrict__ x,
                                               const float* __restrict__ g,
                                               const float* __restrict__ bta,
                                               _Float16* __restrict__ xn) {
  __shared__ float s_sum[256], s_sq[256];
  const int bc = blockIdx.x;          // b*64 + c
  const int c  = bc & 63;
  const float* row = x + (size_t)bc * Nn;
  float s = 0.f, q = 0.f;
  for (int i = threadIdx.x; i < Nn; i += 256) { float v = row[i]; s += v; q += v * v; }
  s_sum[threadIdx.x] = s; s_sq[threadIdx.x] = q;
  __syncthreads();
  for (int off = 128; off > 0; off >>= 1) {
    if (threadIdx.x < off) {
      s_sum[threadIdx.x] += s_sum[threadIdx.x + off];
      s_sq[threadIdx.x]  += s_sq[threadIdx.x + off];
    }
    __syncthreads();
  }
  const float mean = s_sum[0] * (1.0f / Nn);
  const float var  = s_sq[0] * (1.0f / Nn) - mean * mean;
  const float rstd = rsqrtf(var + 1e-5f);
  const float gm = g[c] * rstd, bt = bta[c];
  _Float16* orow = xn + (size_t)bc * Nn;
  for (int i = threadIdx.x; i < Nn; i += 256)
    orow[i] = (_Float16)((row[i] - mean) * gm + bt);
}

// ---------------------------------------------------------------------------
// Kernel 2: Q,K,V projections via WMMA.  Q pre-scaled by 1/sqrt(D).
// Q,K: f16 [B][N][D] (token-major).  V: f16 [B][D][N] (d-major, so the
// attention kernel's V B-fragments are contiguous in LDS).
// ---------------------------------------------------------------------------
__global__ __launch_bounds__(256) void k_qkv(const _Float16* __restrict__ xn,
    const float* __restrict__ wq, const float* __restrict__ bq,
    const float* __restrict__ wk, const float* __restrict__ bk,
    const float* __restrict__ wv, const float* __restrict__ bv,
    _Float16* __restrict__ Qb, _Float16* __restrict__ Kb, _Float16* __restrict__ Vt) {
  __shared__ _Float16 sWq[64 * 64], sWk[64 * 64], sWv[64 * 64];
  const int tid = threadIdx.x, wave = tid >> 5, lane = tid & 31;
  const int col = lane & 15, hi = (lane >> 4) & 1;
  const int b = blockIdx.y;
  const int nbase = blockIdx.x * 128 + wave * 16;

  for (int i = tid; i < 4096; i += 256) {
    sWq[i] = (_Float16)wq[i]; sWk[i] = (_Float16)wk[i]; sWv[i] = (_Float16)wv[i];
  }
  __syncthreads();

  // A fragments of xn^T: rows = 16 tokens, K-dim = channels (two chunks of 32)
  v16h a0, a1;
  const int n = nbase + col;
  #pragma unroll
  for (int e = 0; e < 16; ++e) {
    int kk = (hi ? 16 : 0) + e;
    a0[e] = xn[((size_t)b * 64 + kk)      * Nn + n];
    a1[e] = xn[((size_t)b * 64 + 32 + kk) * Nn + n];
  }

  #pragma unroll
  for (int t = 0; t < 4; ++t) {
    const int dg = t * 16 + col;           // output feature (B-fragment column)
    v16h bq0, bq1, bk0, bk1, bv0, bv1;
    #pragma unroll
    for (int e = 0; e < 16; ++e) {
      int c0 = (hi ? 16 : 0) + e, c1 = 32 + c0;   // B: K-dim = channel
      bq0[e] = sWq[dg * 64 + c0]; bq1[e] = sWq[dg * 64 + c1];
      bk0[e] = sWk[dg * 64 + c0]; bk1[e] = sWk[dg * 64 + c1];
      bv0[e] = sWv[dg * 64 + c0]; bv1[e] = sWv[dg * 64 + c1];
    }
    v8f aq = {}, ak = {}, av = {};
    aq = WMMA_F32_F16(a0, bq0, aq); aq = WMMA_F32_F16(a1, bq1, aq);
    ak = WMMA_F32_F16(a0, bk0, ak); ak = WMMA_F32_F16(a1, bk1, ak);
    av = WMMA_F32_F16(a0, bv0, av); av = WMMA_F32_F16(a1, bv1, av);
    const float bqv = bq[dg], bkv = bk[dg], bvv = bv[dg];
    #pragma unroll
    for (int r = 0; r < 8; ++r) {
      const size_t nrow = (size_t)b * Nn + nbase + r + 8 * hi;
      Qb[nrow * 64 + dg] = (_Float16)((aq[r] + bqv) * 0.125f);  // 1/sqrt(64)
      Kb[nrow * 64 + dg] = (_Float16)(ak[r] + bkv);
      Vt[((size_t)b * 64 + dg) * Nn + (nbase + r + 8 * hi)] = (_Float16)(av[r] + bvv);
    }
  }
}

// ---------------------------------------------------------------------------
// Kernel 3: flash attention (online softmax) + fused output projection.
// Block = 8 waves = 128 queries; wave owns 16 queries x all 64 head dims.
// K/V tiles double-buffered in LDS, filled by TDM (wave 0), overlapped
// with the WMMA + softmax of the current tile.
// ---------------------------------------------------------------------------
__global__ __launch_bounds__(256) void k_attn(const _Float16* __restrict__ Qb,
    const _Float16* __restrict__ Kb, const _Float16* __restrict__ Vt,
    const float* __restrict__ wo, const float* __restrict__ bo,
    float* __restrict__ out) {
  __shared__ _Float16 sK[2][64 * 64];    // 2 x 8 KB key tile   [key][d]
  __shared__ _Float16 sV[2][64 * 64];    // 2 x 8 KB value tile [d][key]
  __shared__ _Float16 sW[64 * 64];       // 8 KB wo (f16)       [e][d]
  __shared__ _Float16 sP[8 * 16 * 64];   // 16 KB per-wave P / O scratch
  const int tid = threadIdx.x, wave = tid >> 5, lane = tid & 31;
  const int col = lane & 15, hi = (lane >> 4) & 1;
  const int b = blockIdx.y;
  const int qbase = blockIdx.x * 128 + wave * 16;

  for (int i = tid; i < 4096; i += 256) sW[i] = (_Float16)wo[i];

  const _Float16* Kbase = Kb + (size_t)b * Nn * 64;  // [key][d]
  const _Float16* Vbase = Vt + (size_t)b * 64 * Nn;  // [d][n]

#if USE_TDM
  auto issue_tile = [&](int kb, int buf) {
    // K tile: one contiguous 4096-element "row" (64 keys x 64 d).
    tdm_load_2d_f16((unsigned)(unsigned long long)&sK[buf][0],
                    Kbase + (size_t)kb * 64 * 64,
                    /*dim0=*/4096, /*dim1=*/64, /*tile0=*/4096, /*tile1=*/1,
                    /*stride0=*/4096);
    // V tile: 64 d-rows of 64 keys, row stride N (strided 2-D tile).
    tdm_load_2d_f16((unsigned)(unsigned long long)&sV[buf][0],
                    Vbase + (size_t)kb * 64,
                    /*dim0=*/4096, /*dim1=*/64, /*tile0=*/64, /*tile1=*/64,
                    /*stride0=*/4096);
  };
#endif

  // Q A-fragments for this wave's 16 queries (two K-chunks over d)
  v16h qf0, qf1;
  {
    const _Float16* Qrow = Qb + ((size_t)b * Nn + qbase + col) * 64;
    #pragma unroll
    for (int e = 0; e < 16; ++e) {
      int kk = (hi ? 16 : 0) + e;
      qf0[e] = Qrow[kk];
      qf1[e] = Qrow[32 + kk];
    }
  }

  v8f O[4]; O[0] = (v8f){}; O[1] = (v8f){}; O[2] = (v8f){}; O[3] = (v8f){};
  float mrow[8], lrow[8];
  #pragma unroll
  for (int r = 0; r < 8; ++r) { mrow[r] = -3.0e38f; lrow[r] = 0.f; }

  _Float16* Pw = sP + wave * (16 * 64);

#if USE_TDM
  if (wave == 0) issue_tile(0, 0);
#endif

  for (int kb = 0; kb < 64; ++kb) {
    const int cur = kb & 1;
#if USE_TDM
    if (wave == 0) {
      if (kb + 1 < 64) {
        issue_tile(kb + 1, cur ^ 1);                // overlap next DMA
        // 2 TDM ops per tile, in-order: <=2 outstanding leaves only "next" pending.
        __builtin_amdgcn_s_wait_tensorcnt(2);
      } else {
        __builtin_amdgcn_s_wait_tensorcnt(0);
      }
    }
#else
    __syncthreads();
    {
      const uint4* k4 = (const uint4*)(Kbase + (size_t)kb * 64 * 64);
      uint4* dk = (uint4*)&sK[cur][0];
      dk[tid] = k4[tid]; dk[tid + 256] = k4[tid + 256];
      // V is d-major: 64 rows of 128 B, strided by N*2 bytes.
      for (int i = tid; i < 512; i += 256) {
        const int d = i >> 3, ko = (i & 7) * 8;
        *(uint4*)&sV[cur][d * 64 + ko] = *(const uint4*)(Vbase + (size_t)d * Nn + kb * 64 + ko);
      }
    }
#endif
    __syncthreads();
    const _Float16* sKc = &sK[cur][0];
    const _Float16* sVc = &sV[cur][0];

    // S = Q * K^T : 4 tiles of 16 keys, each accumulating 2 K-chunks of d
    v8f S[4];
    #pragma unroll
    for (int t = 0; t < 4; ++t) {
      v16h bk0, bk1;
      const int keyrow = (t * 16 + col) * 64;   // B column n = key
      #pragma unroll
      for (int e = 0; e < 16; ++e) {
        int d0 = (hi ? 16 : 0) + e;             // B K-dim = d (contiguous)
        bk0[e] = sKc[keyrow + d0];
        bk1[e] = sKc[keyrow + 32 + d0];
      }
      v8f acc = {};
      acc = WMMA_F32_F16(qf0, bk0, acc);
      acc = WMMA_F32_F16(qf1, bk1, acc);
      S[t] = acc;
    }

    // Online softmax update (row stats via 16-lane shuffle reductions)
    float alpha[8], mnew[8];
    #pragma unroll
    for (int r = 0; r < 8; ++r) {
      float mt = fmaxf(fmaxf(S[0][r], S[1][r]), fmaxf(S[2][r], S[3][r]));
      mt = red16_max(mt);
      float mn = fmaxf(mrow[r], mt);
      alpha[r] = __expf(mrow[r] - mn);
      mrow[r] = mn; mnew[r] = mn;
    }
    float ls[8];
    #pragma unroll
    for (int r = 0; r < 8; ++r) ls[r] = 0.f;
    #pragma unroll
    for (int t = 0; t < 4; ++t) {
      #pragma unroll
      for (int r = 0; r < 8; ++r) {
        float p = __expf(S[t][r] - mnew[r]);
        ls[r] += p;
        Pw[(r + 8 * hi) * 64 + t * 16 + col] = (_Float16)p;  // D-layout -> [row][key]
      }
    }
    #pragma unroll
    for (int r = 0; r < 8; ++r) {
      lrow[r] = lrow[r] * alpha[r] + red16_sum(ls[r]);
      O[0][r] *= alpha[r]; O[1][r] *= alpha[r]; O[2][r] *= alpha[r]; O[3][r] *= alpha[r];
    }
    __syncthreads();

    // Re-read P as A-fragments (transpose through LDS), then O += P * V
    v16h ap0, ap1;
    #pragma unroll
    for (int e = 0; e < 16; ++e) {
      int kk = (hi ? 16 : 0) + e;
      ap0[e] = Pw[col * 64 + kk];
      ap1[e] = Pw[col * 64 + 32 + kk];
    }
    #pragma unroll
    for (int t = 0; t < 4; ++t) {
      v16h bv0, bv1;
      const int drow = (t * 16 + col) * 64;       // B column n = d; V is [d][key]
      #pragma unroll
      for (int e = 0; e < 16; ++e) {
        int k0 = (hi ? 16 : 0) + e;               // B K-dim = key (contiguous)
        bv0[e] = sVc[drow + k0];
        bv1[e] = sVc[drow + 32 + k0];
      }
      O[t] = WMMA_F32_F16(ap0, bv0, O[t]);
      O[t] = WMMA_F32_F16(ap1, bv1, O[t]);
    }
    __syncthreads();   // all reads of buf `cur` done before it is refilled
  }

  // Normalize, then fuse the output projection: out = wo * att + bo
  #pragma unroll
  for (int r = 0; r < 8; ++r) {
    float inv = 1.0f / lrow[r];
    O[0][r] *= inv; O[1][r] *= inv; O[2][r] *= inv; O[3][r] *= inv;
  }
  #pragma unroll
  for (int r = 0; r < 8; ++r) {
    int rr = (r + 8 * hi) * 64;
    Pw[rr + 0 * 16 + col] = (_Float16)O[0][r];
    Pw[rr + 1 * 16 + col] = (_Float16)O[1][r];
    Pw[rr + 2 * 16 + col] = (_Float16)O[2][r];
    Pw[rr + 3 * 16 + col] = (_Float16)O[3][r];
  }
  __syncthreads();
  v16h ao0, ao1;
  #pragma unroll
  for (int e = 0; e < 16; ++e) {
    int kk = (hi ? 16 : 0) + e;
    ao0[e] = Pw[col * 64 + kk];
    ao1[e] = Pw[col * 64 + 32 + kk];
  }
  #pragma unroll
  for (int t = 0; t < 4; ++t) {
    v16h bw0, bw1;
    const int erow = (t * 16 + col) * 64;   // B column n = output feature e
    #pragma unroll
    for (int e = 0; e < 16; ++e) {
      int d0 = (hi ? 16 : 0) + e;           // B K-dim = d (contiguous)
      bw0[e] = sW[erow + d0];
      bw1[e] = sW[erow + 32 + d0];
    }
    v8f acc = {};
    acc = WMMA_F32_F16(ao0, bw0, acc);
    acc = WMMA_F32_F16(ao1, bw1, acc);
    const int eg = t * 16 + col;
    const float bov = bo[eg];
    #pragma unroll
    for (int r = 0; r < 8; ++r) {
      const int n = qbase + r + 8 * hi;
      out[((size_t)(b * 64 + eg)) * Nn + n] = acc[r] + bov;   // [B, D, N]
    }
  }
}

// ---------------------------------------------------------------------------
extern "C" void kernel_launch(void* const* d_in, const int* in_sizes, int n_in,
                              void* d_out, int out_size, void* d_ws, size_t ws_size,
                              hipStream_t stream) {
  (void)in_sizes; (void)n_in; (void)out_size; (void)ws_size;
  const float* x  = (const float*)d_in[0];
  const float* gw = (const float*)d_in[1];
  const float* gb = (const float*)d_in[2];
  const float* wq = (const float*)d_in[3];
  const float* bq = (const float*)d_in[4];
  const float* wk = (const float*)d_in[5];
  const float* bk = (const float*)d_in[6];
  const float* wv = (const float*)d_in[7];
  const float* bv = (const float*)d_in[8];
  const float* wo = (const float*)d_in[9];
  const float* bo = (const float*)d_in[10];
  float* out = (float*)d_out;

  // Workspace: xn + Q + K + V(d-major), all f16 (8 MB total), rewritten per call.
  _Float16* xn = (_Float16*)d_ws;
  _Float16* Qb = xn + (size_t)Bn * Cn * Nn;
  _Float16* Kb = Qb + (size_t)Bn * Nn * Dn;
  _Float16* Vt = Kb + (size_t)Bn * Nn * Dn;

  k_inorm<<<dim3(Bn * Cn), 256, 0, stream>>>(x, gw, gb, xn);
  k_qkv  <<<dim3(32, Bn), 256, 0, stream>>>(xn, wq, bq, wk, bk, wv, bv, Qb, Kb, Vt);
  k_attn <<<dim3(32, Bn), 256, 0, stream>>>(Qb, Kb, Vt, wo, bo, out);
}
